// EnhanceMHSA_v3_75204877353102
// MI455X (gfx1250) — matrix operand order repaired and checked by reading, not compiled, not run on hardware
//
#include <hip/hip_runtime.h>
#include <math.h>

typedef _Float16 h16  __attribute__((ext_vector_type(16)));
typedef _Float16 h8_t __attribute__((ext_vector_type(8)));
typedef float    f8_t __attribute__((ext_vector_type(8)));
typedef unsigned int u32x4 __attribute__((ext_vector_type(4)));
typedef int          i32x4 __attribute__((ext_vector_type(4)));
typedef int          i32x8 __attribute__((ext_vector_type(8)));

#define HEADS 4
#define DK    32
#define DV    32
#define CH    64
#define HID   256
#define NB    8
#define HW    4096   /* 64*64 query tokens per batch */
#define KVTOK 1024   /* 32*32 kv tokens per batch */
#define MTOT  (NB*HW)   /* 32768 */
#define MKV   (NB*KVTOK)/* 8192  */

// ---------------------------------------------------------------------------
// WMMA fragment loaders, following CDNA5 ISA VGPR layouts (CDNA5 ISA):
//  A 16x32 f16 : lanes 0-15 -> M=lane,    halves = K {0..7, 16..23}
//                lanes16-31 -> M=lane-16, halves = K {8..15, 24..31}
//  B 32x16 f16 : lanes 0-15 -> N=lane,    halves = K {0..7, 16..23}
//                lanes16-31 -> N=lane-16, halves = K {8..15, 24..31}
//  C 16x16 f32 : VGPR r -> row (r + 8*(lane>=16)), lane&15 -> col
// ---------------------------------------------------------------------------
__device__ __forceinline__ h16 load_A16x32(const _Float16* base, int ld) {
  int lane = threadIdx.x & 31;
  int grp  = lane >> 4;
  int m    = lane & 15;
  const _Float16* p = base + (size_t)m * ld + grp * 8;
  h8_t lo = *(const h8_t*)(p);
  h8_t hi = *(const h8_t*)(p + 16);
  h16 r;
#pragma unroll
  for (int i = 0; i < 8; ++i) { r[i] = lo[i]; r[i + 8] = hi[i]; }
  return r;
}

__device__ __forceinline__ h16 load_B16x32(const _Float16* base, int ld) {
  int lane = threadIdx.x & 31;
  int grp  = lane >> 4;
  int n    = lane & 15;
  const _Float16* p = base + (size_t)n * ld + grp * 8;
  h8_t lo = *(const h8_t*)(p);
  h8_t hi = *(const h8_t*)(p + 16);
  h16 r;
#pragma unroll
  for (int i = 0; i < 8; ++i) { r[i] = lo[i]; r[i + 8] = hi[i]; }
  return r;
}

__device__ __forceinline__ f8_t wmma_f16f32(h16 a, h16 b, f8_t c) {
  return __builtin_amdgcn_wmma_f32_16x16x32_f16(false, a, false, b, (short)0, c,
                                                false, false);
}

// ---------------------------------------------------------------------------
// TDM: issue a 2-D tensor_load_to_lds of a [rows x Kdim] f16 tile starting at
// `gptr` (row stride = Kdim elements) into LDS offset 0.  D# layout per
// CDNA5 ISA.  Caller must be wave-uniform.
// The asm escape afterwards tells the compiler that LDS pointed to by `ldsp`
// has been written (the builtin is not modeled as writing the shared array;
// without this, loads from it are undef-folded and deleted).
// ---------------------------------------------------------------------------
__device__ __forceinline__ void tdm_load_tile_f16(const _Float16* gptr,
                                                  _Float16* ldsp,
                                                  int Kdim, int rows) {
  unsigned long long ga = (unsigned long long)(const void*)gptr;
  u32x4 g0;
  g0[0] = 1u;                                             // count=1, user mode
  g0[1] = 0u;                                             // lds_addr = 0
  g0[2] = (unsigned)ga;                                   // global_addr lo32
  g0[3] = ((unsigned)(ga >> 32) & 0x01FFFFFFu) | 0x80000000u; // addr hi + type=2
  i32x8 g1;
  g1[0] = (int)(1u << 16);                                // data_size = 2 bytes
  g1[1] = (int)(((unsigned)Kdim & 0xFFFFu) << 16);        // tensor_dim0 lo16
  g1[2] = (int)(((unsigned)Kdim >> 16) |
                (((unsigned)rows & 0xFFFFu) << 16));      // td0 hi16 | td1 lo16
  g1[3] = (int)(((unsigned)rows >> 16) |
                ((unsigned)Kdim << 16));                  // td1 hi16 | tile_dim0
  g1[4] = rows;                                           // tile_dim1 (tile_dim2=0)
  g1[5] = Kdim;                                           // tensor_dim0_stride lo32
  g1[6] = 0;
  g1[7] = 0;
  i32x4 gz = {0, 0, 0, 0};
#if defined(__clang_major__) && (__clang_major__ >= 23)
  i32x8 gz8 = {0, 0, 0, 0, 0, 0, 0, 0};
  __builtin_amdgcn_tensor_load_to_lds(g0, g1, gz, gz, gz8, 0);
#else
  __builtin_amdgcn_tensor_load_to_lds(g0, g1, gz, gz, 0);
#endif
  __builtin_amdgcn_s_wait_tensorcnt(0);
  asm volatile("" : : "v"(ldsp) : "memory");  // TDM wrote LDS: publish it
}

// ---------------------------------------------------------------------------
// Small prep kernels
// ---------------------------------------------------------------------------
__global__ void cvt_f16_kernel(const float* __restrict__ src,
                               _Float16* __restrict__ dst, int n) {
  int i = blockIdx.x * blockDim.x + threadIdx.x;
  if (i < n) dst[i] = (_Float16)src[i];
}

__global__ void bnfold_kernel(const float* __restrict__ g, const float* __restrict__ b,
                              const float* __restrict__ m, const float* __restrict__ v,
                              float* __restrict__ s, float* __restrict__ t, int n) {
  int i = blockIdx.x * blockDim.x + threadIdx.x;
  if (i < n) {
    float sc = g[i] * rsqrtf(v[i] + 1e-5f);
    s[i] = sc;
    t[i] = b[i] - m[i] * sc;
  }
}

// ---------------------------------------------------------------------------
// LayerNorm over channels + NCHW->token-major f16 transpose (raw + normed)
// ---------------------------------------------------------------------------
__global__ void ln_kernel(const float* __restrict__ x, const float* __restrict__ g,
                          const float* __restrict__ be, _Float16* __restrict__ xt,
                          _Float16* __restrict__ xn) {
  int tid = blockIdx.x * blockDim.x + threadIdx.x;
  if (tid >= MTOT) return;
  int b = tid >> 12, t = tid & 4095;
  const float* xp = x + (size_t)b * (CH * HW) + t;
  float vals[CH];
  float s = 0.f, s2 = 0.f;
#pragma unroll
  for (int c = 0; c < CH; ++c) {
    float v = xp[(size_t)c * HW];
    vals[c] = v; s += v; s2 += v * v;
  }
  float mean = s * (1.f / CH);
  float var  = s2 * (1.f / CH) - mean * mean;
  float inv  = rsqrtf(var + 1e-5f);
  _Float16* pt = xt + (size_t)tid * CH;
  _Float16* pn = xn + (size_t)tid * CH;
#pragma unroll
  for (int c = 0; c < CH; ++c) {
    pt[c] = (_Float16)vals[c];
    pn[c] = (_Float16)((vals[c] - mean) * inv * g[c] + be[c]);
  }
}

// ---------------------------------------------------------------------------
// Depthwise 3x3 stride-2 pad-1 on NHWC f16 + folded BN + PReLU
// ---------------------------------------------------------------------------
__global__ void dwconv_kernel(const _Float16* __restrict__ h1, const float* __restrict__ w,
                              const float* __restrict__ s, const float* __restrict__ t,
                              const float* __restrict__ a, _Float16* __restrict__ h2) {
  int c  = threadIdx.x;                 // 0..255 channel
  int bk = blockIdx.x;                  // b*1024 + output token
  int b  = bk >> 10, ot = bk & 1023;
  int oy = ot >> 5, ox = ot & 31;
  float acc = 0.f;
#pragma unroll
  for (int ky = 0; ky < 3; ++ky) {
    int iy = 2 * oy - 1 + ky;
    if (iy < 0 || iy >= 64) continue;
#pragma unroll
    for (int kx = 0; kx < 3; ++kx) {
      int ix = 2 * ox - 1 + kx;
      if (ix < 0 || ix >= 64) continue;
      acc += (float)h1[((size_t)b * HW + iy * 64 + ix) * HID + c] *
             w[c * 9 + ky * 3 + kx];
    }
  }
  float y = s[c] * acc + t[c];
  y = y < 0.f ? a[c] * y : y;
  h2[((size_t)b * KVTOK + ot) * HID + c] = (_Float16)y;
}

// ---------------------------------------------------------------------------
// WMMA GEMM:  C[M,N] = A[M,K](f16 rowmajor) * W[N,K](f16 rowmajor)^T
// Block = 128 threads = 4 waves.  Block tile 64x64:
//   - wave 0 TDM-stages the 64xK weight tile into LDS (one tensor_load_to_lds),
//     s_wait_tensorcnt, then workgroup barrier
//   - each wave owns 16 rows; per k-step one A fragment feeds 4 WMMAs against
//     the 4 n-subtiles read from LDS (A reuse x4, B from LDS)
// Epilogue modes:
//  0: +bias(e0), f16 rowmajor            1: BN(e0,e1)+PReLU(e2), f16 rowmajor
//  2: BN(e0,e1), f16 rowmajor            3: +bias, per-head [b,H,rpb,32] layout
//  4: +bias, transposed-V [b,H,32,rpb]   5: +bias +residual, f32 rowmajor (d_out)
// ---------------------------------------------------------------------------
__global__ void gemm_wmma_kernel(const _Float16* __restrict__ A,
                                 const _Float16* __restrict__ W,
                                 int Mdim, int Ndim, int Kdim, int mode, int rpb,
                                 const float* __restrict__ e0,
                                 const float* __restrict__ e1,
                                 const float* __restrict__ e2,
                                 const float* __restrict__ resid,
                                 _Float16* __restrict__ outh,
                                 float* __restrict__ outf) {
  __shared__ __align__(16) _Float16 wlds[64 * 256];   // up to K=256 (32 KB)
  int wave = threadIdx.x >> 5;
  int n0 = blockIdx.x * 64;
  int m0 = blockIdx.y * 64 + wave * 16;

  if (wave == 0)
    tdm_load_tile_f16(W + (size_t)n0 * Kdim, wlds, Kdim, 64);
  __syncthreads();
  asm volatile("" : : "v"((_Float16*)wlds) : "memory");  // anchor LDS contents

  f8_t acc0 = {}, acc1 = {}, acc2 = {}, acc3 = {};
  for (int k0 = 0; k0 < Kdim; k0 += 32) {
    h16 a = load_A16x32(A + (size_t)m0 * Kdim + k0, Kdim);
    h16 b0 = load_B16x32(wlds + (size_t)0  * Kdim + k0, Kdim);
    h16 b1 = load_B16x32(wlds + (size_t)16 * Kdim + k0, Kdim);
    h16 b2 = load_B16x32(wlds + (size_t)32 * Kdim + k0, Kdim);
    h16 b3 = load_B16x32(wlds + (size_t)48 * Kdim + k0, Kdim);
    acc0 = wmma_f16f32(a, b0, acc0);
    acc1 = wmma_f16f32(a, b1, acc1);
    acc2 = wmma_f16f32(a, b2, acc2);
    acc3 = wmma_f16f32(a, b3, acc3);
  }

  int lane = threadIdx.x & 31;
  int grp = lane >> 4, nn = lane & 15;
#pragma unroll
  for (int t4 = 0; t4 < 4; ++t4) {
    f8_t acc = (t4 == 0) ? acc0 : (t4 == 1) ? acc1 : (t4 == 2) ? acc2 : acc3;
    int col = n0 + t4 * 16 + nn;
#pragma unroll
    for (int r = 0; r < 8; ++r) {
      int row = m0 + r + grp * 8;
      float v = acc[r];
      if (mode == 0) {
        outh[(size_t)row * Ndim + col] = (_Float16)(v + e0[col]);
      } else if (mode == 1) {
        float y = e0[col] * v + e1[col];
        y = y < 0.f ? e2[col] * y : y;
        outh[(size_t)row * Ndim + col] = (_Float16)y;
      } else if (mode == 2) {
        outh[(size_t)row * Ndim + col] = (_Float16)(e0[col] * v + e1[col]);
      } else if (mode == 3) {
        int b = row / rpb, t = row - b * rpb;
        int h = col >> 5, d = col & 31;
        outh[(((size_t)b * HEADS + h) * rpb + t) * DK + d] = (_Float16)(v + e0[col]);
      } else if (mode == 4) {
        int b = row / rpb, t = row - b * rpb;
        int h = col >> 5, d = col & 31;
        outh[(((size_t)b * HEADS + h) * DV + d) * (size_t)rpb + t] =
            (_Float16)(v + e0[col]);
      } else {
        size_t p = (size_t)row * CH + col;
        outf[p] = v + e0[col] + resid[p];
      }
    }
  }
  (void)Mdim;
}

// ---------------------------------------------------------------------------
// Flash-style bias-augmented attention.
// grid (64 qtiles, HEADS, NB); block = 128 threads = 4 waves, 16 q-rows/wave.
// Per 32-KV chunk: S = Q*K^T (2 WMMAs), scale+Bbias, online softmax,
// P staged via per-wave LDS tile to re-layout C->A, then O += P*V (2 WMMAs).
// ---------------------------------------------------------------------------
__global__ void attn_kernel(const _Float16* __restrict__ Q,
                            const _Float16* __restrict__ Km,
                            const _Float16* __restrict__ Vt,
                            const float* __restrict__ Bbias,
                            _Float16* __restrict__ aout) {
  __shared__ __align__(16) _Float16 plds[4 * 16 * 32];
  int wave = threadIdx.x >> 5;
  int lane = threadIdx.x & 31;
  int grp = lane >> 4, nn = lane & 15;
  int h = blockIdx.y, b = blockIdx.z;
  int m0 = blockIdx.x * 64 + wave * 16;

  const _Float16* Qb = Q  + (((size_t)b * HEADS + h) * HW + m0) * DK;
  const _Float16* Kb = Km + (((size_t)b * HEADS + h) * KVTOK) * DK;
  const _Float16* Vb = Vt + (((size_t)b * HEADS + h) * DV) * (size_t)KVTOK;
  const float* bias  = Bbias + ((size_t)h * HW + m0 + grp * 8) * KVTOK;

  h16 qa = load_A16x32(Qb, DK);           // DK==32 -> single A fragment
  f8_t acc0 = {}, acc1 = {};
  float mrun[8], lrun[8];
#pragma unroll
  for (int r = 0; r < 8; ++r) { mrun[r] = -1e30f; lrun[r] = 0.f; }

  _Float16* pl = plds + wave * (16 * 32);
  const float scale = 0.17677669529663688f; // 32^-0.5

  for (int j = 0; j < KVTOK; j += 32) {
    h16 kb0 = load_B16x32(Kb + (size_t)j * DK, DK);
    h16 kb1 = load_B16x32(Kb + (size_t)(j + 16) * DK, DK);
    f8_t z = {};
    f8_t s0 = wmma_f16f32(qa, kb0, z);
    f8_t s1 = wmma_f16f32(qa, kb1, z);
    if (j + 32 < KVTOK)                       // pull next bias chunk toward L2/L0
      __builtin_prefetch((const void*)(bias + j + 32), 0, 1);

    float mx[8];
#pragma unroll
    for (int r = 0; r < 8; ++r) {
      const float* bp = bias + (size_t)r * KVTOK + j;
      s0[r] = s0[r] * scale + bp[nn];
      s1[r] = s1[r] * scale + bp[16 + nn];
      mx[r] = fmaxf(s0[r], s1[r]);
    }
#pragma unroll
    for (int off = 8; off >= 1; off >>= 1)
#pragma unroll
      for (int r = 0; r < 8; ++r)
        mx[r] = fmaxf(mx[r], __shfl_xor(mx[r], off, 32));

    float rs[8];
#pragma unroll
    for (int r = 0; r < 8; ++r) {
      float nm = fmaxf(mrun[r], mx[r]);
      float al = __expf(mrun[r] - nm);
      mrun[r] = nm;
      acc0[r] *= al; acc1[r] *= al; lrun[r] *= al;
      s0[r] = __expf(s0[r] - nm);
      s1[r] = __expf(s1[r] - nm);
      rs[r] = s0[r] + s1[r];
    }
#pragma unroll
    for (int off = 8; off >= 1; off >>= 1)
#pragma unroll
      for (int r = 0; r < 8; ++r)
        rs[r] += __shfl_xor(rs[r], off, 32);

#pragma unroll
    for (int r = 0; r < 8; ++r) {
      lrun[r] += rs[r];
      pl[(r + grp * 8) * 32 + nn]      = (_Float16)s0[r];
      pl[(r + grp * 8) * 32 + 16 + nn] = (_Float16)s1[r];
    }
    asm volatile("s_wait_dscnt 0" ::: "memory");
    h16 pa  = load_A16x32(pl, 32);                       // C->A relayout via LDS
    h16 vb0 = load_B16x32(Vb + j, KVTOK);                // dv 0..15
    h16 vb1 = load_B16x32(Vb + (size_t)16 * KVTOK + j, KVTOK); // dv 16..31
    acc0 = wmma_f16f32(pa, vb0, acc0);
    acc1 = wmma_f16f32(pa, vb1, acc1);
  }

#pragma unroll
  for (int r = 0; r < 8; ++r) {
    float inv = 1.f / lrun[r];
    int row = m0 + r + grp * 8;
    size_t base = ((size_t)b * HW + row) * (HEADS * DV) + (size_t)h * DV;
    aout[base + nn]      = (_Float16)(acc0[r] * inv);
    aout[base + 16 + nn] = (_Float16)(acc1[r] * inv);
  }
}

// ---------------------------------------------------------------------------
// Host orchestration
// ---------------------------------------------------------------------------
extern "C" void kernel_launch(void* const* d_in, const int* in_sizes, int n_in,
                              void* d_out, int out_size, void* d_ws, size_t ws_size,
                              hipStream_t stream) {
  (void)in_sizes; (void)n_in; (void)out_size; (void)ws_size;
  const float* x     = (const float*)d_in[0];
  const float* ln_g  = (const float*)d_in[1];
  const float* ln_b  = (const float*)d_in[2];
  const float* wq    = (const float*)d_in[3];
  const float* bq    = (const float*)d_in[4];
  const float* wk    = (const float*)d_in[5];
  const float* bk    = (const float*)d_in[6];
  const float* wv    = (const float*)d_in[7];
  const float* bv    = (const float*)d_in[8];
  const float* wo    = (const float*)d_in[9];
  const float* bo    = (const float*)d_in[10];
  const float* Bbias = (const float*)d_in[11];
  const float* kw1  = (const float*)d_in[12];
  const float* kb1g = (const float*)d_in[13]; const float* kb1b = (const float*)d_in[14];
  const float* kb1m = (const float*)d_in[15]; const float* kb1v = (const float*)d_in[16];
  const float* ka1  = (const float*)d_in[17];
  const float* kwdw = (const float*)d_in[18];
  const float* kb2g = (const float*)d_in[19]; const float* kb2b = (const float*)d_in[20];
  const float* kb2m = (const float*)d_in[21]; const float* kb2v = (const float*)d_in[22];
  const float* ka2  = (const float*)d_in[23];
  const float* kw3  = (const float*)d_in[24];
  const float* kb3g = (const float*)d_in[25]; const float* kb3b = (const float*)d_in[26];
  const float* kb3m = (const float*)d_in[27]; const float* kb3v = (const float*)d_in[28];
  const float* vw1  = (const float*)d_in[29];
  const float* vb1g = (const float*)d_in[30]; const float* vb1b = (const float*)d_in[31];
  const float* vb1m = (const float*)d_in[32]; const float* vb1v = (const float*)d_in[33];
  const float* va1  = (const float*)d_in[34];
  const float* vwdw = (const float*)d_in[35];
  const float* vb2g = (const float*)d_in[36]; const float* vb2b = (const float*)d_in[37];
  const float* vb2m = (const float*)d_in[38]; const float* vb2v = (const float*)d_in[39];
  const float* va2  = (const float*)d_in[40];
  const float* vw3  = (const float*)d_in[41];
  const float* vb3g = (const float*)d_in[42]; const float* vb3b = (const float*)d_in[43];
  const float* vb3m = (const float*)d_in[44]; const float* vb3v = (const float*)d_in[45];

  char* ws = (char*)d_ws;
  size_t off = 0;
  auto carve = [&](size_t bytes) -> void* {
    void* p = ws + off;
    off += (bytes + 255) & ~(size_t)255;
    return p;
  };
  _Float16* wq16  = (_Float16*)carve(8192 * 2);
  _Float16* wk16  = (_Float16*)carve(8192 * 2);
  _Float16* wv16  = (_Float16*)carve(8192 * 2);
  _Float16* wo16  = (_Float16*)carve(8192 * 2);
  _Float16* w1k16 = (_Float16*)carve(16384 * 2);
  _Float16* w3k16 = (_Float16*)carve(16384 * 2);
  _Float16* w1v16 = (_Float16*)carve(16384 * 2);
  _Float16* w3v16 = (_Float16*)carve(16384 * 2);
  float* s1k = (float*)carve(256 * 4); float* t1k = (float*)carve(256 * 4);
  float* s2k = (float*)carve(256 * 4); float* t2k = (float*)carve(256 * 4);
  float* s3k = (float*)carve(64 * 4);  float* t3k = (float*)carve(64 * 4);
  float* s1v = (float*)carve(256 * 4); float* t1v = (float*)carve(256 * 4);
  float* s2v = (float*)carve(256 * 4); float* t2v = (float*)carve(256 * 4);
  float* s3v = (float*)carve(64 * 4);  float* t3v = (float*)carve(64 * 4);
  _Float16* xt   = (_Float16*)carve((size_t)MTOT * CH * 2);
  _Float16* xn   = (_Float16*)carve((size_t)MTOT * CH * 2);
  _Float16* Qbuf = (_Float16*)carve((size_t)MTOT * 128 * 2);
  _Float16* h1   = (_Float16*)carve((size_t)MTOT * HID * 2);
  _Float16* h2   = (_Float16*)carve((size_t)MKV * HID * 2);
  _Float16* kf   = (_Float16*)carve((size_t)MKV * CH * 2);
  _Float16* Km   = (_Float16*)carve((size_t)MKV * 128 * 2);
  _Float16* VtB  = (_Float16*)carve((size_t)MKV * 128 * 2);
  _Float16* aout = (_Float16*)carve((size_t)MTOT * 128 * 2);

  auto cvt = [&](const float* s, _Float16* d, int n) {
    cvt_f16_kernel<<<(n + 255) / 256, 256, 0, stream>>>(s, d, n);
  };
  cvt(wq, wq16, 8192);  cvt(wk, wk16, 8192);
  cvt(wv, wv16, 8192);  cvt(wo, wo16, 8192);
  cvt(kw1, w1k16, 16384); cvt(kw3, w3k16, 16384);
  cvt(vw1, w1v16, 16384); cvt(vw3, w3v16, 16384);
  bnfold_kernel<<<1, 256, 0, stream>>>(kb1g, kb1b, kb1m, kb1v, s1k, t1k, 256);
  bnfold_kernel<<<1, 256, 0, stream>>>(kb2g, kb2b, kb2m, kb2v, s2k, t2k, 256);
  bnfold_kernel<<<1, 64,  0, stream>>>(kb3g, kb3b, kb3m, kb3v, s3k, t3k, 64);
  bnfold_kernel<<<1, 256, 0, stream>>>(vb1g, vb1b, vb1m, vb1v, s1v, t1v, 256);
  bnfold_kernel<<<1, 256, 0, stream>>>(vb2g, vb2b, vb2m, vb2v, s2v, t2v, 256);
  bnfold_kernel<<<1, 64,  0, stream>>>(vb3g, vb3b, vb3m, vb3v, s3v, t3v, 64);

  // --- LayerNorm + tokenization ---
  ln_kernel<<<MTOT / 256, 256, 0, stream>>>(x, ln_g, ln_b, xt, xn);

  // --- Q projection: [32768,64] x [128,64]^T -> per-head layout ---
  gemm_wmma_kernel<<<dim3(128 / 64, MTOT / 64), 128, 0, stream>>>(
      xn, wq16, MTOT, 128, 64, 3, HW, bq, nullptr, nullptr, nullptr, Qbuf, nullptr);

  // --- K branch: expand -> dw -> proj -> K projection ---
  gemm_wmma_kernel<<<dim3(HID / 64, MTOT / 64), 128, 0, stream>>>(
      xt, w1k16, MTOT, HID, 64, 1, 0, s1k, t1k, ka1, nullptr, h1, nullptr);
  dwconv_kernel<<<NB * KVTOK, HID, 0, stream>>>(h1, kwdw, s2k, t2k, ka2, h2);
  gemm_wmma_kernel<<<dim3(CH / 64, MKV / 64), 128, 0, stream>>>(
      h2, w3k16, MKV, CH, HID, 2, 0, s3k, t3k, nullptr, nullptr, kf, nullptr);
  gemm_wmma_kernel<<<dim3(128 / 64, MKV / 64), 128, 0, stream>>>(
      kf, wk16, MKV, 128, 64, 3, KVTOK, bk, nullptr, nullptr, nullptr, Km, nullptr);

  // --- V branch (reuses h1/h2/kf), V stored transposed [b,H,dv,kv] ---
  gemm_wmma_kernel<<<dim3(HID / 64, MTOT / 64), 128, 0, stream>>>(
      xt, w1v16, MTOT, HID, 64, 1, 0, s1v, t1v, va1, nullptr, h1, nullptr);
  dwconv_kernel<<<NB * KVTOK, HID, 0, stream>>>(h1, vwdw, s2v, t2v, va2, h2);
  gemm_wmma_kernel<<<dim3(CH / 64, MKV / 64), 128, 0, stream>>>(
      h2, w3v16, MKV, CH, HID, 2, 0, s3v, t3v, nullptr, nullptr, kf, nullptr);
  gemm_wmma_kernel<<<dim3(128 / 64, MKV / 64), 128, 0, stream>>>(
      kf, wv16, MKV, 128, 64, 4, KVTOK, bv, nullptr, nullptr, nullptr, VtB, nullptr);

  // --- attention ---
  attn_kernel<<<dim3(HW / 64, HEADS, NB), 128, 0, stream>>>(Qbuf, Km, VtB, Bbias, aout);

  // --- output projection + residual (raw-view flat add) -> d_out f32 ---
  gemm_wmma_kernel<<<dim3(CH / 64, MTOT / 64), 128, 0, stream>>>(
      aout, wo16, MTOT, CH, 128, 5, 0, bo, nullptr, nullptr, x, nullptr,
      (float*)d_out);
}
